// ARMTApproach_36610301231698
// MI455X (gfx1250) — matrix operand order repaired and checked by reading, not compiled, not run on hardware
//
#include <hip/hip_runtime.h>
#include <hip/hip_bf16.h>
#include <math.h>

// ---------------------------------------------------------------------------
// ARMT associative-memory layer for MI455X (gfx1250, wave32, WMMA).
//   f32->bf16 pack -> 5 projection GEMMs (bf16 WMMA, fused bias/sigmoid)
//   -> dpfp+L2norm features -> per-head WMMA GEMMs against augmented memory
//   (extra column 128 carries the norm / coef / den reductions) -> epilogues.
// GEMM staging: double-buffered GLOBAL_LOAD_ASYNC_TO_LDS_B128 (ASYNCcnt) when
// the toolchain exposes it, else global_load_b128 -> ds_store_b128.
// ---------------------------------------------------------------------------

typedef __bf16 bf16_t;
typedef __attribute__((ext_vector_type(16))) __bf16 v16bf;
typedef __attribute__((ext_vector_type(8)))  __bf16 v8bf;
typedef __attribute__((ext_vector_type(4)))  __bf16 v4bf;
typedef __attribute__((ext_vector_type(8)))  float  v8f;

#if defined(__has_builtin)
#  if __has_builtin(__builtin_amdgcn_global_load_async_to_lds_b128) && \
      __has_builtin(__builtin_amdgcn_s_wait_asynccnt)
#    define USE_ASYNC_COPY 1
#  endif
#endif
#ifndef USE_ASYNC_COPY
#  define USE_ASYNC_COPY 0
#endif

#if USE_ASYNC_COPY
typedef __attribute__((__vector_size__(4 * sizeof(int)))) int v4i;
typedef __attribute__((address_space(1))) v4i as1_v4i;   // global b128
typedef __attribute__((address_space(3))) v4i as3_v4i;   // LDS b128
#endif

static constexpr int S_LEN = 4096;
static constexpr int HID   = 2048;
static constexpr int NH    = 16;
static constexpr int NHKV  = 4;
static constexpr int DHEAD = 128;
static constexpr int FDIM  = 768;   // 2*3*128
static constexpr int NAUG  = 144;   // 128 data cols + col128 reduction + pad to x16
static constexpr int NAUGP = 192;   // BT row padding to a multiple of 64 (guard-free staging)

// ---------------------------------------------------------------------------
// Shared WMMA micro-kernel: one wave's 32x32 quadrant, one 32-wide K step.
// A fragment 16x32: lane row = wm+i*16+(lane&15), k = halfL*8+{0..7} / 16+halfL*8+{0..7}
// B fragment 32x16: lane col = wn+j*16+(lane&15), k = halfL*8+{0..7} / 16+halfL*8+{0..7}
// ---------------------------------------------------------------------------
__device__ __forceinline__ void wmma_accum(const bf16_t (&tA)[64][40],
                                           const bf16_t (&tB)[64][40],
                                           int wm, int wn, int halfL, int l15,
                                           v8f acc[2][2])
{
#pragma unroll
    for (int i = 0; i < 2; ++i) {
        const bf16_t* ar = &tA[wm + i * 16 + l15][0];
        v8bf a0 = *(const v8bf*)(ar + halfL * 8);
        v8bf a1 = *(const v8bf*)(ar + 16 + halfL * 8);
        v16bf afrag = __builtin_shufflevector(a0, a1, 0, 1, 2, 3, 4, 5, 6, 7,
                                              8, 9, 10, 11, 12, 13, 14, 15);
#pragma unroll
        for (int j = 0; j < 2; ++j) {
            const bf16_t* br = &tB[wn + j * 16 + l15][0];
            v8bf b0 = *(const v8bf*)(br + halfL * 8);
            v8bf b1 = *(const v8bf*)(br + 16 + halfL * 8);
            v16bf bfrag = __builtin_shufflevector(b0, b1, 0, 1, 2, 3, 4, 5, 6, 7,
                                                  8, 9, 10, 11, 12, 13, 14, 15);
            acc[i][j] = __builtin_amdgcn_wmma_f32_16x16x32_bf16(
                false, afrag, false, bfrag, (short)0, acc[i][j], false, false);
        }
    }
}

// ---------------------------------------------------------------------------
// bf16 WMMA GEMM:  C[M,N] = act( A[M,K] x BT[N,K]^T + bias )
// Requirements: M % 64 == 0, K % 32 == 0, BT has >= ceil(N/64)*64 rows.
// Block: 128 threads = 4 waves, 64x64 tile; wave owns a 32x32 quadrant.
// ---------------------------------------------------------------------------
__global__ __launch_bounds__(128)
void gemm_bf16_wmma(const bf16_t* __restrict__ A, long lda,
                    const bf16_t* __restrict__ BT, long ldb,
                    const float* __restrict__ bias,
                    float* __restrict__ Cf, bf16_t* __restrict__ Cb, int ldc,
                    int M, int N, int K, int act)
{
    __shared__ bf16_t lA[2][64][40];   // 32 data + 8 pad (keeps 16B alignment)
    __shared__ bf16_t lB[2][64][40];

    const int tid  = threadIdx.x;
    const int wave = tid >> 5;
    const int lane = tid & 31;
    const int m0 = blockIdx.y * 64;
    const int n0 = blockIdx.x * 64;
    const int wm = (wave >> 1) * 32;
    const int wn = (wave & 1) * 32;
    const int halfL = lane >> 4;
    const int l15   = lane & 15;

    v8f acc[2][2] = {};

#if USE_ASYNC_COPY
    // ---- double-buffered async global->LDS staging (ASYNCcnt) ----
    auto stage = [&](int kk, int buf) {
#pragma unroll
        for (int it = 0; it < 2; ++it) {
            int i = tid + it * 128;            // 0..255
            int row = i >> 2, g = i & 3;       // 4 x 8-element (16B) groups per row
            __builtin_amdgcn_global_load_async_to_lds_b128(
                (as1_v4i*)(A + (long)(m0 + row) * lda + kk + g * 8),
                (as3_v4i*)&lA[buf][row][g * 8], 0, 0);
            __builtin_amdgcn_global_load_async_to_lds_b128(
                (as1_v4i*)(BT + (long)(n0 + row) * ldb + kk + g * 8),
                (as3_v4i*)&lB[buf][row][g * 8], 0, 0);
        }
    };
    stage(0, 0);
    __builtin_amdgcn_s_wait_asynccnt(0);
    __syncthreads();
    int buf = 0;
    for (int kk = 0; kk < K; kk += 32) {
        if (kk + 32 < K) stage(kk + 32, buf ^ 1);      // prefetch next tile
        wmma_accum(lA[buf], lB[buf], wm, wn, halfL, l15, acc);
        __builtin_amdgcn_s_wait_asynccnt(0);
        __syncthreads();
        buf ^= 1;
    }
#else
    // ---- fallback: VGPR-staged b128 copies ----
    for (int kk = 0; kk < K; kk += 32) {
#pragma unroll
        for (int it = 0; it < 2; ++it) {
            int i = tid + it * 128;
            int row = i >> 2, g = i & 3;
            *(v8bf*)&lA[0][row][g * 8] =
                *(const v8bf*)(A  + (long)(m0 + row) * lda + kk + g * 8);
            *(v8bf*)&lB[0][row][g * 8] =
                *(const v8bf*)(BT + (long)(n0 + row) * ldb + kk + g * 8);
        }
        __syncthreads();
        if (tid == 0 && kk + 32 < K) {
            __builtin_prefetch((const void*)(A  + (long)m0 * lda + kk + 32), 0, 1);
            __builtin_prefetch((const void*)(BT + (long)n0 * ldb + kk + 32), 0, 1);
        }
        wmma_accum(lA[0], lB[0], wm, wn, halfL, l15, acc);
        __syncthreads();
    }
#endif

    // C layout: N = lane&15; M = r + 8*(lane>=16) for accumulator element r
#pragma unroll
    for (int i = 0; i < 2; ++i)
#pragma unroll
        for (int j = 0; j < 2; ++j) {
            int n = n0 + wn + j * 16 + l15;
            if (n < N) {
                float bv = bias ? bias[n] : 0.f;
#pragma unroll
                for (int r = 0; r < 8; ++r) {
                    int m = m0 + wm + i * 16 + halfL * 8 + r;
                    float val = acc[i][j][r] + bv;
                    if (act == 1) val = 1.f / (1.f + __expf(-val));
                    if (Cf) Cf[(long)m * ldc + n] = val;
                    else    Cb[(long)m * ldc + n] = (bf16_t)val;
                }
            }
        }
}

// ---------------------------------------------------------------------------
// dpfp feature map + L2 normalize. One wave per (head,row).
// x2 = [relu(x), relu(-x)] staged in LDS; feat[(j-1)*256+i] = x2[i]*x2[(i-j)%256], j=1..3.
// ---------------------------------------------------------------------------
__global__ __launch_bounds__(256)
void dpfp_features(const bf16_t* __restrict__ X,  // [S, Hn*128]
                   bf16_t* __restrict__ F,        // [Hn, S, 768]
                   int S, int Hn)
{
    __shared__ float x2s[8][256];
    const int wave = threadIdx.x >> 5;
    const int lane = threadIdx.x & 31;
    const long row = (long)blockIdx.x * 8 + wave;   // grid sized exactly Hn*S/8
    const int h = (int)(row / S);
    const int s = (int)(row % S);

    const bf16_t* x = X + (long)s * (Hn * 128) + h * 128;
    float* x2 = x2s[wave];
#pragma unroll
    for (int t = 0; t < 4; ++t) {
        float v = (float)x[lane + 32 * t];
        x2[lane + 32 * t]       = v > 0.f ? v : 0.f;
        x2[128 + lane + 32 * t] = v < 0.f ? -v : 0.f;
    }
    __syncthreads();

    float feats[24];
    float ss = 0.f;
#pragma unroll
    for (int t = 0; t < 24; ++t) {
        int f = lane + 32 * t;
        int j = (f >> 8) + 1;
        int i = f & 255;
        float v = x2[i] * x2[(i - j) & 255];
        feats[t] = v;
        ss += v * v;
    }
#pragma unroll
    for (int off = 16; off > 0; off >>= 1)
        ss += __shfl_xor(ss, off, 32);
    float scale = 1.f / fmaxf(sqrtf(ss), 1e-12f);

    bf16_t* out = F + ((long)h * S + s) * FDIM;
#pragma unroll
    for (int t = 0; t < 24; ++t)
        out[lane + 32 * t] = (bf16_t)(feats[t] * scale);
}

// ---------------------------------------------------------------------------
// 32x32 LDS tile transpose: src [R][C] -> dst [C][R]  (per blockIdx.z batch)
// ---------------------------------------------------------------------------
__global__ __launch_bounds__(256)
void transpose_bf16(const bf16_t* __restrict__ src, bf16_t* __restrict__ dst,
                    int R, int C)
{
    __shared__ bf16_t tile[32][36];   // pad keeps 8B alignment, dodges conflicts
    const long base = (long)blockIdx.z * R * C;
    const int r0 = blockIdx.y * 32, c0 = blockIdx.x * 32;
    const int r  = threadIdx.x >> 3;
    const int cq = (threadIdx.x & 7) * 4;

    *(v4bf*)&tile[r][cq] = *(const v4bf*)(src + base + (long)(r0 + r) * C + c0 + cq);
    __syncthreads();

    v4bf o;
#pragma unroll
    for (int e = 0; e < 4; ++e) o[e] = tile[cq + e][r];
    *(v4bf*)(dst + base + (long)(c0 + r) * R + r0 + cq) = o;
}

// ---------------------------------------------------------------------------
// Elementwise kernels
// ---------------------------------------------------------------------------
__global__ void cvt_f32_bf16(const float* __restrict__ src, bf16_t* __restrict__ dst, long n)
{
    long i = (long)blockIdx.x * blockDim.x + threadIdx.x;
    if (i < n) dst[i] = (bf16_t)src[i];
}

// memAT[h][c][f] (c in [0,192)): c<128 -> memory[h][f][c], c==128 -> norm[h][f], else 0
__global__ void build_memAT(const float* __restrict__ mem, const float* __restrict__ nrm,
                            bf16_t* __restrict__ memAT)
{
    long i = (long)blockIdx.x * blockDim.x + threadIdx.x;
    if (i >= (long)NH * NAUGP * FDIM) return;
    int f = (int)(i % FDIM);
    int c = (int)((i / FDIM) % NAUGP);
    int h = (int)(i / ((long)FDIM * NAUGP));
    float v = 0.f;
    if (c < 128)       v = mem[((long)h * FDIM + f) * DHEAD + c];
    else if (c == 128) v = nrm[(long)h * FDIM + f];
    memAT[i] = (bf16_t)v;
}

// wmvT[h][c][s] (c in [0,192)): c<128 -> (v - num/denom)*mb, c==128 -> coef, else 0
__global__ void make_wmvT(const float* __restrict__ num, const bf16_t* __restrict__ v,
                          const bf16_t* __restrict__ mb, bf16_t* __restrict__ wmvT)
{
    long i = (long)blockIdx.x * blockDim.x + threadIdx.x;
    if (i >= (long)NH * NAUGP * S_LEN) return;
    int s = (int)(i % S_LEN);
    int c = (int)((i / S_LEN) % NAUGP);
    int h = (int)(i / ((long)S_LEN * NAUGP));
    long hs = (long)h * S_LEN + s;
    float denom = num[hs * NAUG + 128] + 1e-8f;
    float out = 0.f;
    if (c < 128) {
        float prev = num[hs * NAUG + c] / denom;
        float vv  = (float)v[(long)s * (NHKV * DHEAD) + (h >> 2) * DHEAD + c];
        float mbv = (float)mb[(long)s * HID + h * DHEAD + c];
        out = (vv - prev) * mbv;
    } else if (c == 128) {
        // ||mk||^2 == 1 after L2 normalization -> coef = clip(1 - denom, 0, 1)
        out = fminf(fmaxf(1.f - denom, 0.f), 1.f);
    }
    wmvT[i] = (bf16_t)out;
}

// new_memory/new_norm (f32 to d_out) + transposed bf16 copy for the assoc GEMM
__global__ void update_mem(const float* __restrict__ mem, const float* __restrict__ nrm,
                           const float* __restrict__ delta,   // [NH][FDIM][NAUG]
                           float* __restrict__ out_mem, float* __restrict__ out_nrm,
                           bf16_t* __restrict__ memAnT)       // [NH][NAUGP][FDIM]
{
    long i = (long)blockIdx.x * blockDim.x + threadIdx.x;
    if (i >= (long)NH * NAUGP * FDIM) return;
    int f = (int)(i % FDIM);
    int c = (int)((i / FDIM) % NAUGP);
    int h = (int)(i / ((long)FDIM * NAUGP));
    long hf = (long)h * FDIM + f;
    float o = 0.f;
    if (c < 128) {
        o = mem[hf * DHEAD + c] + delta[hf * NAUG + c];
        out_mem[hf * DHEAD + c] = o;
    } else if (c == 128) {
        o = nrm[hf] + delta[hf * NAUG + c];
        out_nrm[hf] = o;
    }
    memAnT[i] = (bf16_t)o;
}

// out = hs + gate * (num_a / den_a)
__global__ void final_out(const float* __restrict__ hs, const bf16_t* __restrict__ gate,
                          const float* __restrict__ numa, float* __restrict__ out)
{
    long i = (long)blockIdx.x * blockDim.x + threadIdx.x;
    if (i >= (long)S_LEN * HID) return;
    int c = (int)(i % HID);
    long s = i / HID;
    int h = c >> 7, d = c & 127;
    long hsidx = (long)h * S_LEN + s;
    float den = numa[hsidx * NAUG + 128] + 1e-8f;
    float mh  = numa[hsidx * NAUG + d] / den;
    out[i] = hs[i] + (float)gate[i] * mh;
}

// ---------------------------------------------------------------------------
// Host-side orchestration
// ---------------------------------------------------------------------------
static inline dim3 grid1d(long n, int bs) { return dim3((unsigned)((n + bs - 1) / bs)); }

extern "C" void kernel_launch(void* const* d_in, const int* in_sizes, int n_in,
                              void* d_out, int out_size, void* d_ws, size_t ws_size,
                              hipStream_t stream)
{
    const float* hs  = (const float*)d_in[0];
    const float* Wq  = (const float*)d_in[1];
    const float* bq  = (const float*)d_in[2];
    const float* Wk  = (const float*)d_in[3];
    const float* bk  = (const float*)d_in[4];
    const float* Wv  = (const float*)d_in[5];
    const float* bv  = (const float*)d_in[6];
    const float* Wg  = (const float*)d_in[7];
    const float* bg  = (const float*)d_in[8];
    const float* Wmb = (const float*)d_in[9];
    const float* bmb = (const float*)d_in[10];
    const float* mem = (const float*)d_in[11];
    const float* nrm = (const float*)d_in[12];

    float* out      = (float*)d_out;                     // [4096, 2048]
    float* out_mem  = out + (long)S_LEN * HID;           // [16, 768, 128]
    float* out_nrm  = out_mem + (long)NH * FDIM * DHEAD; // [16, 768]

    // workspace carve-up (256B aligned)
    char* wp = (char*)d_ws;
    auto carve = [&](size_t bytes) { char* p = wp; wp += (bytes + 255) & ~(size_t)255; return p; };
    bf16_t* hs_bf  = (bf16_t*)carve((size_t)S_LEN * HID * 2);
    bf16_t* Wq_bf  = (bf16_t*)carve((size_t)HID * HID * 2);
    bf16_t* Wk_bf  = (bf16_t*)carve((size_t)NHKV * DHEAD * HID * 2);
    bf16_t* Wv_bf  = (bf16_t*)carve((size_t)NHKV * DHEAD * HID * 2);
    bf16_t* Wg_bf  = (bf16_t*)carve((size_t)HID * HID * 2);
    bf16_t* Wmb_bf = (bf16_t*)carve((size_t)HID * HID * 2);
    bf16_t* q_bf   = (bf16_t*)carve((size_t)S_LEN * HID * 2);
    bf16_t* k_bf   = (bf16_t*)carve((size_t)S_LEN * NHKV * DHEAD * 2);
    bf16_t* v_bf   = (bf16_t*)carve((size_t)S_LEN * NHKV * DHEAD * 2);
    bf16_t* g_bf   = (bf16_t*)carve((size_t)S_LEN * HID * 2);
    bf16_t* mb_bf  = (bf16_t*)carve((size_t)S_LEN * HID * 2);
    bf16_t* mk     = (bf16_t*)carve((size_t)NHKV * S_LEN * FDIM * 2);  // [4][S][F]
    bf16_t* mkT    = (bf16_t*)carve((size_t)NHKV * FDIM * S_LEN * 2);  // [4][F][S]
    bf16_t* mq     = (bf16_t*)carve((size_t)NH * S_LEN * FDIM * 2);    // [16][S][F]
    bf16_t* memAT  = (bf16_t*)carve((size_t)NH * NAUGP * FDIM * 2);    // [16][192][768]
    bf16_t* memAnT = (bf16_t*)carve((size_t)NH * NAUGP * FDIM * 2);
    float*  numb   = (float*)carve((size_t)NH * S_LEN * NAUG * 4);     // [16][S][144]
    bf16_t* wmvT   = (bf16_t*)carve((size_t)NH * NAUGP * S_LEN * 2);   // [16][192][S]
    float*  delta  = (float*)carve((size_t)NH * FDIM * NAUG * 4);      // [16][768][144]
    float*  numa   = (float*)carve((size_t)NH * S_LEN * NAUG * 4);

    // 1) pack inputs to bf16
    cvt_f32_bf16<<<grid1d((long)S_LEN * HID, 256), 256, 0, stream>>>(hs, hs_bf, (long)S_LEN * HID);
    cvt_f32_bf16<<<grid1d((long)HID * HID, 256), 256, 0, stream>>>(Wq, Wq_bf, (long)HID * HID);
    cvt_f32_bf16<<<grid1d((long)NHKV * DHEAD * HID, 256), 256, 0, stream>>>(Wk, Wk_bf, (long)NHKV * DHEAD * HID);
    cvt_f32_bf16<<<grid1d((long)NHKV * DHEAD * HID, 256), 256, 0, stream>>>(Wv, Wv_bf, (long)NHKV * DHEAD * HID);
    cvt_f32_bf16<<<grid1d((long)HID * HID, 256), 256, 0, stream>>>(Wg, Wg_bf, (long)HID * HID);
    cvt_f32_bf16<<<grid1d((long)HID * HID, 256), 256, 0, stream>>>(Wmb, Wmb_bf, (long)HID * HID);
    build_memAT<<<grid1d((long)NH * NAUGP * FDIM, 256), 256, 0, stream>>>(mem, nrm, memAT);

    // 2) projections: C = act(hs_bf @ W^T + b); W [N][K] serves directly as BT
    auto proj = [&](const bf16_t* W, const float* bias, bf16_t* C, int N, int act) {
        dim3 g((unsigned)(N / 64), (unsigned)(S_LEN / 64));
        gemm_bf16_wmma<<<g, 128, 0, stream>>>(hs_bf, (long)HID, W, (long)HID,
                                              bias, nullptr, C, N,
                                              S_LEN, N, HID, act);
    };
    proj(Wq_bf,  bq,  q_bf,  HID, 0);
    proj(Wk_bf,  bk,  k_bf,  NHKV * DHEAD, 0);
    proj(Wv_bf,  bv,  v_bf,  NHKV * DHEAD, 0);
    proj(Wg_bf,  bg,  g_bf,  HID, 1);
    proj(Wmb_bf, bmb, mb_bf, HID, 1);

    // 3) dpfp feature maps (one wave per row), then transpose mk for the delta GEMM
    dpfp_features<<<dim3((unsigned)((long)NHKV * S_LEN / 8)), 256, 0, stream>>>(k_bf, mk, S_LEN, NHKV);
    dpfp_features<<<dim3((unsigned)((long)NH * S_LEN / 8)), 256, 0, stream>>>(q_bf, mq, S_LEN, NH);
    {
        dim3 g(FDIM / 32, S_LEN / 32, NHKV);
        transpose_bf16<<<g, 256, 0, stream>>>(mk, mkT, S_LEN, FDIM);
    }

    // 4) num[h] = mk[h/4] @ memAT[h]^T   (col 128 = denom)
    for (int h = 0; h < NH; ++h) {
        dim3 g((NAUG + 63) / 64, S_LEN / 64);
        gemm_bf16_wmma<<<g, 128, 0, stream>>>(
            mk + (long)(h >> 2) * S_LEN * FDIM, (long)FDIM,
            memAT + (long)h * NAUGP * FDIM, (long)FDIM,
            nullptr, numb + (long)h * S_LEN * NAUG, nullptr, NAUG,
            S_LEN, NAUG, FDIM, 0);
    }

    // 5) weighted mv, written transposed (+ coef in row 128)
    make_wmvT<<<grid1d((long)NH * NAUGP * S_LEN, 256), 256, 0, stream>>>(numb, v_bf, mb_bf, wmvT);

    // 6) delta[h] = mkT[h/4] @ wmvT[h]^T  (col 128 = delta_norm)
    for (int h = 0; h < NH; ++h) {
        dim3 g((NAUG + 63) / 64, FDIM / 64);
        gemm_bf16_wmma<<<g, 128, 0, stream>>>(
            mkT + (long)(h >> 2) * FDIM * S_LEN, (long)S_LEN,
            wmvT + (long)h * NAUGP * S_LEN, (long)S_LEN,
            nullptr, delta + (long)h * FDIM * NAUG, nullptr, NAUG,
            FDIM, NAUG, S_LEN, 0);
    }

    // 7) memory/norm update (f32 outputs + transposed bf16 copy)
    update_mem<<<grid1d((long)NH * NAUGP * FDIM, 256), 256, 0, stream>>>(
        mem, nrm, delta, out_mem, out_nrm, memAnT);

    // 8) numa[h] = mq[h] @ memAnT[h]^T  (col 128 = den_a)
    for (int h = 0; h < NH; ++h) {
        dim3 g((NAUG + 63) / 64, S_LEN / 64);
        gemm_bf16_wmma<<<g, 128, 0, stream>>>(
            mq + (long)h * S_LEN * FDIM, (long)FDIM,
            memAnT + (long)h * NAUGP * FDIM, (long)FDIM,
            nullptr, numa + (long)h * S_LEN * NAUG, nullptr, NAUG,
            S_LEN, NAUG, FDIM, 0);
    }

    // 9) residual + gate
    final_out<<<grid1d((long)S_LEN * HID, 256), 256, 0, stream>>>(hs, g_bf, numa, out);
}
